// Conv2d_35270271434780
// MI455X (gfx1250) — matrix operand order repaired and checked by reading, not compiled, not run on hardware
//
#include <hip/hip_runtime.h>

// CDNA5 / gfx1250 fused StyleGAN2 upsample(x2)+3x3 conv+FIR+bias.
// Phase-decomposed implicit GEMM on v_wmma_f32_16x16x32_bf16 (wave32),
// async global->LDS staging of the bf16 input tile.
//
// d_ws layout: [0, 32MB) x in bf16 ; [32MB, 36.7MB) packed phase weights.

typedef __attribute__((ext_vector_type(16))) __bf16 bf16x16;
typedef __attribute__((ext_vector_type(8)))  float  f32x8;

#define ASYNC_STAGE 1

// Per-axis effective-kernel coefficient tables: CTAB[phase][t+1][a]
__device__ __constant__ float CTAB[2][3][3] = {
  { {0.00f, 0.25f, 0.75f},
    {0.75f, 0.75f, 0.25f},
    {0.25f, 0.00f, 0.00f} },
  { {0.00f, 0.00f, 0.25f},
    {0.25f, 0.75f, 0.75f},
    {0.75f, 0.25f, 0.00f} }
};

static __device__ __forceinline__ unsigned short f2bf(float f) {
  unsigned int u = __float_as_uint(f);
  u += 0x7fffu + ((u >> 16) & 1u);   // round-to-nearest-even
  return (unsigned short)(u >> 16);
}

// ---------------------------------------------------------------------------
// Pass 0: convert x (f32) -> xbf (bf16), 8 elements per thread.
// ---------------------------------------------------------------------------
__global__ void __launch_bounds__(256)
cvt_x_bf16(const float* __restrict__ x, unsigned short* __restrict__ xbf) {
  size_t i = ((size_t)blockIdx.x * 256 + threadIdx.x) * 8;
  float4 a = *(const float4*)(x + i);
  float4 b = *(const float4*)(x + i + 4);
  unsigned short h[8] = { f2bf(a.x), f2bf(a.y), f2bf(a.z), f2bf(a.w),
                          f2bf(b.x), f2bf(b.y), f2bf(b.z), f2bf(b.w) };
  *(uint4*)(xbf + i) = *(const uint4*)h;
}

// ---------------------------------------------------------------------------
// Pass 1: pack Weff[phase][tap][ci][co] into WMMA A-fragment order (bf16):
//   off = ((((phase*9+tap)*8+kb)*16+mb)*32+lane)*16 + h
//   A-layout (ISA 7.12.2, 16-bit A 16x32): lane = m + 16*g,
//   per-lane K index: kk = (h&7) + 8*g + 16*(h>>3)
// ---------------------------------------------------------------------------
__global__ void __launch_bounds__(256)
prep_phase_weights(const float* __restrict__ w,
                   unsigned short* __restrict__ wpack) {
  int t = blockIdx.x * 256 + threadIdx.x;   // [0, 147456)
  int lane  = t & 31;
  int mb    = (t >> 5) & 15;
  int kb    = (t >> 9) & 7;
  int pt    = t >> 12;                      // phase*9 + tap, [0,36)
  int tap   = pt % 9;
  int phase = pt / 9;
  int py = phase >> 1, px = phase & 1;
  int tyi = tap / 3, txi = tap % 3;
  int co = mb * 16 + (lane & 15);
  int g  = lane >> 4;

  unsigned short vals[16];
#pragma unroll
  for (int h = 0; h < 16; ++h) {
    int kk = (h & 7) + 8 * g + 16 * (h >> 3);
    int ci = kb * 32 + kk;
    float acc = 0.f;
#pragma unroll
    for (int ay = 0; ay < 3; ++ay) {
      float cy = CTAB[py][tyi][ay];
#pragma unroll
      for (int ax = 0; ax < 3; ++ax) {
        float cx = CTAB[px][txi][ax];
        acc += cy * cx * w[(((ay * 3 + ax) * 256) + ci) * 256 + co];
      }
    }
    vals[h] = f2bf(acc);
  }
  uint4* dst = (uint4*)(wpack + (size_t)t * 16);
  dst[0] = *(const uint4*)&vals[0];
  dst[1] = *(const uint4*)&vals[8];
}

// ---------------------------------------------------------------------------
// Pass 2: one block = (batch n, 8x8 pixel tile), computes ALL 4 phases and
// all 256 output channels. Per phase: GEMM M=256(co), N=64(pix), K=2304.
// Waves partition M only (8 waves x 32 co) -> zero A duplication from L2.
// ---------------------------------------------------------------------------
__global__ void __launch_bounds__(256)
upconv_wmma(const unsigned short* __restrict__ xbf,
            const unsigned short* __restrict__ wpack,
            const float* __restrict__ bias,
            float* __restrict__ out) {
  __shared__ __align__(16) unsigned short ldsX[10 * 10 * 256]; // 51200 B bf16

  const int n  = blockIdx.y;
  const int R0 = (blockIdx.x >> 3) * 8;   // input-row tile origin
  const int S0 = (blockIdx.x & 7) * 8;    // input-col tile origin

  // ---- Stage zero-padded 10x10x256 bf16 tile into LDS (16B chunks) ----
  for (int idx = threadIdx.x; idx < 3200; idx += 256) {
    int ci8 = idx & 31;          // channel/8
    int cr  = idx >> 5;          // [0,100)
    int col = cr % 10, row = cr / 10;
    int xr = R0 - 1 + row, xc = S0 - 1 + col;
    unsigned loff = (unsigned)(row * 10 + col) * 256u + (unsigned)ci8 * 8u;
    if ((unsigned)xr < 64u && (unsigned)xc < 64u) {
      const unsigned short* gp =
          xbf + ((((size_t)n * 64 + xr) * 64 + xc) * 256 + ci8 * 8);
#if ASYNC_STAGE
      // LDS byte offset = low 32 bits of the generic LDS pointer.
      unsigned ldsOff = (unsigned)(size_t)&ldsX[loff];
      unsigned long long gaddr = (unsigned long long)(size_t)gp;
      asm volatile("global_load_async_to_lds_b128 %0, %1, off"
                   :: "v"(ldsOff), "v"(gaddr) : "memory");
#else
      *(uint4*)&ldsX[loff] = *(const uint4*)gp;
#endif
    } else {
      *(uint4*)&ldsX[loff] = make_uint4(0u, 0u, 0u, 0u);  // zero halo
    }
  }
#if ASYNC_STAGE
#if __has_builtin(__builtin_amdgcn_s_wait_asynccnt)
  __builtin_amdgcn_s_wait_asynccnt(0);
#else
  asm volatile("s_wait_asynccnt 0x0" ::: "memory");
#endif
#endif
  __syncthreads();

  const int lane = threadIdx.x & 31;
  const int wave = threadIdx.x >> 5;     // 8 waves over M (co): 32 rows each
  const int g    = lane >> 4;

  // B-fragment per-lane bases (B layout: lane = n%16 + 16*g; per-lane
  // K(ci) = 8*g + s (s<8), 16 + 8*g + (s-8) (s>=8) => two 16B LDS chunks).
  int pbase[4];
#pragma unroll
  for (int nf = 0; nf < 4; ++nf) {
    int pix = nf * 16 + (lane & 15);
    pbase[nf] = ((pix >> 3) * 10 + (pix & 7)) * 256 + 8 * g;
  }

  // Bias for this wave's two co-16 blocks (co = wave*32 + mf*16 + 8*g + i)
  float4 bl[2], bh[2];
#pragma unroll
  for (int mf = 0; mf < 2; ++mf) {
    int co = wave * 32 + mf * 16 + 8 * g;
    bl[mf] = *(const float4*)(bias + co);
    bh[mf] = *(const float4*)(bias + co + 4);
  }

  union BF { bf16x16 v; uint4 q[2]; };

  for (int phase = 0; phase < 4; ++phase) {
    const int py = phase >> 1, px = phase & 1;

    f32x8 acc[2][4];
#pragma unroll
    for (int mf = 0; mf < 2; ++mf)
#pragma unroll
      for (int nf = 0; nf < 4; ++nf)
        acc[mf][nf] = (f32x8){0.f, 0.f, 0.f, 0.f, 0.f, 0.f, 0.f, 0.f};

    // A base into packed weights: mb = wave*2 + mf
    const size_t abase = (size_t)phase * 589824 + (size_t)(wave * 2) * 512
                       + (size_t)lane * 16;

    for (int tap = 0; tap < 9; ++tap) {
      const int tyi = tap / 3, txi = tap % 3;
      const int boff = (tyi * 10 + txi) * 256;
      const unsigned short* atap = wpack + abase + (size_t)tap * 65536;
#pragma unroll 2
      for (int kb = 0; kb < 8; ++kb) {
        BF b0, b1, b2, b3;
        b0.q[0] = *(const uint4*)&ldsX[pbase[0] + boff + kb * 32];
        b0.q[1] = *(const uint4*)&ldsX[pbase[0] + boff + kb * 32 + 16];
        b1.q[0] = *(const uint4*)&ldsX[pbase[1] + boff + kb * 32];
        b1.q[1] = *(const uint4*)&ldsX[pbase[1] + boff + kb * 32 + 16];
        b2.q[0] = *(const uint4*)&ldsX[pbase[2] + boff + kb * 32];
        b2.q[1] = *(const uint4*)&ldsX[pbase[2] + boff + kb * 32 + 16];
        b3.q[0] = *(const uint4*)&ldsX[pbase[3] + boff + kb * 32];
        b3.q[1] = *(const uint4*)&ldsX[pbase[3] + boff + kb * 32 + 16];

        const unsigned short* ap = atap + (size_t)kb * 8192;
        bf16x16 a0 = *(const bf16x16*)(ap);
        bf16x16 a1 = *(const bf16x16*)(ap + 512);

        acc[0][0] = __builtin_amdgcn_wmma_f32_16x16x32_bf16(false, a0, false, b0.v, (short)0, acc[0][0], false, false);
        acc[0][1] = __builtin_amdgcn_wmma_f32_16x16x32_bf16(false, a0, false, b1.v, (short)0, acc[0][1], false, false);
        acc[0][2] = __builtin_amdgcn_wmma_f32_16x16x32_bf16(false, a0, false, b2.v, (short)0, acc[0][2], false, false);
        acc[0][3] = __builtin_amdgcn_wmma_f32_16x16x32_bf16(false, a0, false, b3.v, (short)0, acc[0][3], false, false);
        acc[1][0] = __builtin_amdgcn_wmma_f32_16x16x32_bf16(false, a1, false, b0.v, (short)0, acc[1][0], false, false);
        acc[1][1] = __builtin_amdgcn_wmma_f32_16x16x32_bf16(false, a1, false, b1.v, (short)0, acc[1][1], false, false);
        acc[1][2] = __builtin_amdgcn_wmma_f32_16x16x32_bf16(false, a1, false, b2.v, (short)0, acc[1][2], false, false);
        acc[1][3] = __builtin_amdgcn_wmma_f32_16x16x32_bf16(false, a1, false, b3.v, (short)0, acc[1][3], false, false);
      }
      if (tap < 8) __builtin_prefetch(atap + 65536, 0, 0);
    }

    // ---- Epilogue: bias + scatter to interleaved output phase ----
    // C layout: VGPR i, lane L -> (M = i + 8*(L>>4), N = L&15); M = co,
    // so the 8 accumulator lanes are 8 consecutive channels -> float4 x2.
#pragma unroll
    for (int mf = 0; mf < 2; ++mf) {
      const int co = wave * 32 + mf * 16 + 8 * g;
#pragma unroll
      for (int nf = 0; nf < 4; ++nf) {
        int pix = nf * 16 + (lane & 15);
        int pr = pix >> 3, pc = pix & 7;
        int orow = 2 * (R0 + pr) + py;
        int ocol = 2 * (S0 + pc) + px;
        float* op = out + ((((size_t)n * 128 + orow) * 128 + ocol) * 256 + co);
        f32x8 a = acc[mf][nf];
        float4 lo = make_float4(a[0] + bl[mf].x, a[1] + bl[mf].y,
                                a[2] + bl[mf].z, a[3] + bl[mf].w);
        float4 hi = make_float4(a[4] + bh[mf].x, a[5] + bh[mf].y,
                                a[6] + bh[mf].z, a[7] + bh[mf].w);
        *(float4*)op       = lo;
        *(float4*)(op + 4) = hi;
      }
    }
  }
}

// ---------------------------------------------------------------------------
extern "C" void kernel_launch(void* const* d_in, const int* in_sizes, int n_in,
                              void* d_out, int out_size, void* d_ws, size_t ws_size,
                              hipStream_t stream) {
  const float* x = (const float*)d_in[0];   // [16,64,64,256] f32
  const float* w = (const float*)d_in[1];   // [3,3,256,256]  f32
  const float* b = (const float*)d_in[2];   // [256]          f32
  float* out = (float*)d_out;               // [16,128,128,256] f32

  unsigned short* xbf   = (unsigned short*)d_ws;           // 33,554,432 B
  unsigned short* wpack = xbf + 16777216;                  //  4,718,592 B

  cvt_x_bf16<<<8192, 256, 0, stream>>>(x, xbf);
  prep_phase_weights<<<576, 256, 0, stream>>>(w, wpack);

  dim3 grid(64, 16);  // 8x8 pixel tiles, 16 batches (phases looped in-kernel)
  upconv_wmma<<<grid, 256, 0, stream>>>(xbf, wpack, b, out);
}
